// Attention_19018115186763
// MI455X (gfx1250) — matrix operand order repaired and checked by reading, not compiled, not run on hardware
//
#include <hip/hip_runtime.h>
#include <hip/hip_bf16.h>
#include <math.h>

// ---------------------------------------------------------------------------
// Problem constants (B=32, N=257, DIM=512, H=8, D=64, NUM_REL=961)
// ---------------------------------------------------------------------------
#define BATCH   32
#define SEQ     257
#define DIM     512
#define HEADS   8
#define HDIM    64
#define NUMREL  961
#define NPAD    288   // SEQ padded to 9 * 32 (WMMA K steps for P@V)
#define NJ      272   // 17 * 16 j-column tiles actually computed
#define RPAD    976   // NUMREL padded to 61 * 16
#define MROWS   (BATCH * SEQ)        // 8224 = 514 * 16 exactly
#define MT      (MROWS / 16)         // 514

typedef __bf16 bf16;
typedef __attribute__((ext_vector_type(16))) __bf16 v16bf;
typedef __attribute__((ext_vector_type(8)))  float  v8f;

__device__ __forceinline__ v8f wmma_bf16(v16bf a, v16bf b, v8f c) {
  // v_wmma_f32_16x16x32_bf16: D = A(16x32) * B(32x16) + C(16x16 f32)
  return __builtin_amdgcn_wmma_f32_16x16x32_bf16(
      false, a, false, b, (short)0, c, false, false);
}

// A fragment (16x32 bf16, element (m,k) at src[m*ld + k]).
// Lane L: m = L%16, half = L/16. VGPR p holds k = half*16 + p*2, +1.
__device__ __forceinline__ v16bf load_A(const bf16* src, int ld, int lane) {
  int m = lane & 15, half = lane >> 4;
  v16bf a;
#pragma unroll
  for (int p = 0; p < 8; ++p) {
    int k = (half << 4) + (p << 1);
    a[2 * p]     = src[m * ld + k];
    a[2 * p + 1] = src[m * ld + k + 1];
  }
  return a;
}

// B fragment (32x16 bf16), element (k,n) stored n-major: src[n*ld + k].
// Lane L: n = L%16; lanes 0-15 hold k=0..15, lanes 16-31 hold k=16..31.
__device__ __forceinline__ v16bf load_B_nmajor(const bf16* src, int ld, int lane) {
  int n = lane & 15, half = lane >> 4;
  v16bf b;
#pragma unroll
  for (int p = 0; p < 8; ++p) {
    int k = (half << 4) + (p << 1);
    b[2 * p]     = src[n * ld + k];
    b[2 * p + 1] = src[n * ld + k + 1];
  }
  return b;
}

// B fragment (32x16 bf16), element (k,n) stored k-major: src[k*ld + n].
__device__ __forceinline__ v16bf load_B_kmajor(const bf16* src, int ld, int lane) {
  int n = lane & 15, half = lane >> 4;
  v16bf b;
#pragma unroll
  for (int p = 0; p < 8; ++p) {
    int k = (half << 4) + (p << 1);
    b[2 * p]     = src[k * ld + n];
    b[2 * p + 1] = src[(k + 1) * ld + n];
  }
  return b;
}

// ---------------------------------------------------------------------------
// Conversion / init kernels
// ---------------------------------------------------------------------------
__global__ void cvt_f32_bf16(const float* __restrict__ src,
                             bf16* __restrict__ dst, int n) {
  int i = blockIdx.x * blockDim.x + threadIdx.x;
  if (i < n) dst[i] = (bf16)src[i];
}

// dst[n*K + k] = (bf16) src[k*Ncols + n]   (transpose to n-major for B frags)
__global__ void cvt_transpose(const float* __restrict__ src,
                              bf16* __restrict__ dst, int K, int Ncols) {
  int i = blockIdx.x * blockDim.x + threadIdx.x;
  if (i < K * Ncols) {
    int n = i / K, k = i % K;
    dst[i] = (bf16)src[k * Ncols + n];
  }
}

// relh[h][r][d] = rel_emb[r][h*64+d], r padded with zeros up to RPAD
__global__ void cvt_rel(const float* __restrict__ rel_emb,
                        bf16* __restrict__ relh) {
  int i = blockIdx.x * blockDim.x + threadIdx.x;
  int total = HEADS * RPAD * HDIM;
  if (i >= total) return;
  int h = i / (RPAD * HDIM);
  int rem = i % (RPAD * HDIM);
  int r = rem / HDIM, d = rem % HDIM;
  relh[i] = (r < NUMREL) ? (bf16)rel_emb[r * DIM + h * HDIM + d] : (bf16)0.0f;
}

// Zero Q/K/V padding rows 257..287 for every (b,h)
__global__ void zero_pad_qkv(bf16* __restrict__ qb, bf16* __restrict__ kb,
                             bf16* __restrict__ vb) {
  int i = blockIdx.x * blockDim.x + threadIdx.x;
  int total = BATCH * HEADS * (NPAD - SEQ) * HDIM;
  if (i >= total) return;
  int d = i % HDIM;
  int t = i / HDIM;
  int row = SEQ + (t % (NPAD - SEQ));
  int bh = t / (NPAD - SEQ);
  size_t off = ((size_t)bh * NPAD + row) * HDIM + d;
  qb[off] = (bf16)0.0f;
  kb[off] = (bf16)0.0f;
  vb[off] = (bf16)0.0f;
}

// ---------------------------------------------------------------------------
// QKV GEMM: [8224 x 512] @ [512 x 1536] -> Q,K,V bf16 [B,H,NPAD,64]
// one wave per 16x64 output tile (4 accumulators; A fragment reused 4x),
// 16 k-steps * 4 = 64 WMMAs per wave
// ---------------------------------------------------------------------------
__global__ __launch_bounds__(32) void qkv_gemm(
    const bf16* __restrict__ xb,     // [8224,512]
    const bf16* __restrict__ wT,     // [1536,512] n-major
    bf16* __restrict__ qb, bf16* __restrict__ kb, bf16* __restrict__ vb) {
  int mt = blockIdx.x, nt4 = blockIdx.y, lane = threadIdx.x;
  v8f acc0 = {0.f, 0.f, 0.f, 0.f, 0.f, 0.f, 0.f, 0.f};
  v8f acc1 = acc0, acc2 = acc0, acc3 = acc0;
#pragma unroll 2
  for (int kk = 0; kk < 16; ++kk) {
    v16bf a = load_A(xb + (size_t)mt * 16 * DIM + kk * 32, DIM, lane);
    const bf16* wbase = wT + (size_t)nt4 * 64 * DIM + kk * 32;
    v16bf b0 = load_B_nmajor(wbase + (size_t)0 * 16 * DIM, DIM, lane);
    acc0 = wmma_bf16(a, b0, acc0);
    v16bf b1 = load_B_nmajor(wbase + (size_t)1 * 16 * DIM, DIM, lane);
    acc1 = wmma_bf16(a, b1, acc1);
    v16bf b2 = load_B_nmajor(wbase + (size_t)2 * 16 * DIM, DIM, lane);
    acc2 = wmma_bf16(a, b2, acc2);
    v16bf b3 = load_B_nmajor(wbase + (size_t)3 * 16 * DIM, DIM, lane);
    acc3 = wmma_bf16(a, b3, acc3);
  }
  int n = lane & 15, half = lane >> 4;
  v8f accs[4] = {acc0, acc1, acc2, acc3};
#pragma unroll
  for (int t = 0; t < 4; ++t) {
    int col = nt4 * 64 + t * 16 + n;
    int part = col >> 9;          // 0=Q 1=K 2=V
    int wi = col & 511;
    int h = wi >> 6, d = wi & 63;
    bf16* dst = (part == 0) ? qb : (part == 1) ? kb : vb;
#pragma unroll
    for (int r = 0; r < 8; ++r) {
      int m = r + half * 8;
      int gm = mt * 16 + m;
      int b_ = gm / SEQ, nn = gm % SEQ;
      dst[(((size_t)b_ * HEADS + h) * NPAD + nn) * HDIM + d] = (bf16)accs[t][r];
    }
  }
}

// ---------------------------------------------------------------------------
// Fused attention per (b, h, 16-row i tile). 128 threads = 4 waves.
//   proj  = Q_tile @ rel_h^T  (WMMA, 61 n-tiles over 976 rel ids) -> LDS bf16
//   S     = Q_tile @ K^T      (WMMA, 17 j-tiles)                  -> LDS f32
//   S     = (S*scale + proj[rel_pos]) * rel_mul ; mask ; softmax (2-pass, LDS)
//   O     = P @ V             (WMMA, 9 k-steps from LDS bf16 probabilities)
// ---------------------------------------------------------------------------
__global__ __launch_bounds__(128) void attn_fused(
    const bf16* __restrict__ qb, const bf16* __restrict__ kb,
    const bf16* __restrict__ vb, const bf16* __restrict__ relh,
    const int* __restrict__ rel_pos,        // [257,257]
    const float* __restrict__ rel_mul,      // [961,8]
    const unsigned char* __restrict__ mask, // [32,256] bool
    bf16* __restrict__ attn_out) {          // [8224,512]
  __shared__ bf16  proj[16][RPAD];   // 31232 B
  __shared__ float Sm[16][NPAD];     // 18432 B
  __shared__ bf16  Pm[16][NPAD];     //  9216 B
  __shared__ float red[16][8];
  __shared__ float rowmax[16];
  __shared__ float rowsum[16];

  const int itile = blockIdx.x, h = blockIdx.y, b = blockIdx.z;
  const int tid = threadIdx.x, lane = tid & 31, wave = tid >> 5;
  const float scale = 0.04419417382f; // 512^-0.5 (full-dim, as in reference)

  const bf16* Qbh = qb + ((size_t)(b * HEADS + h)) * NPAD * HDIM;
  const bf16* Kbh = kb + ((size_t)(b * HEADS + h)) * NPAD * HDIM;
  const bf16* Vbh = vb + ((size_t)(b * HEADS + h)) * NPAD * HDIM;
  const bf16* relb = relh + (size_t)h * RPAD * HDIM;

  // Q tile A-fragments for k=d chunks [0,32) and [32,64)
  v16bf qa0 = load_A(Qbh + (size_t)itile * 16 * HDIM, HDIM, lane);
  v16bf qa1 = load_A(Qbh + (size_t)itile * 16 * HDIM + 32, HDIM, lane);

  const int n_ = lane & 15, half_ = lane >> 4;

  // ---- Stage 1: proj[i, r] for all 976 rel ids ----
  for (int rt = wave; rt < RPAD / 16; rt += 4) {
    v8f acc = {0.f, 0.f, 0.f, 0.f, 0.f, 0.f, 0.f, 0.f};
    v16bf b0 = load_B_nmajor(relb + (size_t)rt * 16 * HDIM, HDIM, lane);
    acc = wmma_bf16(qa0, b0, acc);
    v16bf b1 = load_B_nmajor(relb + (size_t)rt * 16 * HDIM + 32, HDIM, lane);
    acc = wmma_bf16(qa1, b1, acc);
#pragma unroll
    for (int r = 0; r < 8; ++r)
      proj[r + half_ * 8][rt * 16 + n_] = (bf16)acc[r];
  }

  // ---- Stage 2: raw scores S = Q K^T over 17 j-tiles ----
  for (int jt = wave; jt < NJ / 16; jt += 4) {
    v8f acc = {0.f, 0.f, 0.f, 0.f, 0.f, 0.f, 0.f, 0.f};
    v16bf b0 = load_B_nmajor(Kbh + (size_t)jt * 16 * HDIM, HDIM, lane);
    acc = wmma_bf16(qa0, b0, acc);
    v16bf b1 = load_B_nmajor(Kbh + (size_t)jt * 16 * HDIM + 32, HDIM, lane);
    acc = wmma_bf16(qa1, b1, acc);
#pragma unroll
    for (int r = 0; r < 8; ++r)
      Sm[r + half_ * 8][jt * 16 + n_] = acc[r];
  }
  __syncthreads();

  // ---- Stage 3: scale, additive bias (proj gather), mult bias, mask ----
  const float NEG_INF = -__builtin_inff();
  for (int idx = tid; idx < 16 * NJ; idx += 128) {
    int r = idx / NJ, j = idx % NJ;
    int gi = itile * 16 + r;
    float s;
    if (gi >= SEQ) {
      s = 0.0f;                          // dead row: keep softmax finite
    } else if (j >= SEQ) {
      s = NEG_INF;                       // padded column
    } else {
      bool vi = (gi == 0) || (mask[b * (SEQ - 1) + gi - 1] != 0);
      bool vj = (j == 0) || (mask[b * (SEQ - 1) + j - 1] != 0);
      if (!(vi && vj)) {
        s = NEG_INF;
      } else {
        int rp = rel_pos[gi * SEQ + j];
        float bias = (float)proj[r][rp];
        float mul = rel_mul[rp * HEADS + h];
        s = (Sm[r][j] * scale + bias) * mul;
      }
    }
    Sm[r][j] = s;
  }
  __syncthreads();

  // ---- Stage 4: two-pass softmax over j (row-parallel, 8 threads/row) ----
  {
    int r = tid >> 3, sub = tid & 7;
    float mx = NEG_INF;
    for (int j = sub; j < NJ; j += 8) mx = fmaxf(mx, Sm[r][j]);
    red[r][sub] = mx;
  }
  __syncthreads();
  if (tid < 16) {
    float mx = red[tid][0];
#pragma unroll
    for (int s2 = 1; s2 < 8; ++s2) mx = fmaxf(mx, red[tid][s2]);
    rowmax[tid] = mx;
  }
  __syncthreads();
  {
    int r = tid >> 3, sub = tid & 7;
    float mx = rowmax[r];
    float sum = 0.0f;
    for (int j = sub; j < NJ; j += 8) {
      float p = __expf(Sm[r][j] - mx);
      Pm[r][j] = (bf16)p;
      sum += p;
    }
    red[r][sub] = sum;
  }
  for (int idx = tid; idx < 16 * (NPAD - NJ); idx += 128) {
    int r = idx >> 4;
    Pm[r][NJ + (idx & 15)] = (bf16)0.0f;   // zero K-pad columns for P@V
  }
  __syncthreads();
  if (tid < 16) {
    float s = red[tid][0];
#pragma unroll
    for (int s2 = 1; s2 < 8; ++s2) s += red[tid][s2];
    rowsum[tid] = s;
  }
  __syncthreads();

  // ---- Stage 5: O = P @ V ; wave w owns d-columns [16w, 16w+16) ----
  {
    v8f acc = {0.f, 0.f, 0.f, 0.f, 0.f, 0.f, 0.f, 0.f};
#pragma unroll
    for (int kt = 0; kt < NPAD / 32; ++kt) {   // 9 K-steps of 32
      v16bf a = load_A(&Pm[0][0] + kt * 32, NPAD, lane);
      v16bf bv = load_B_kmajor(Vbh + (size_t)kt * 32 * HDIM + wave * 16,
                               HDIM, lane);
      acc = wmma_bf16(a, bv, acc);
    }
#pragma unroll
    for (int r = 0; r < 8; ++r) {
      int m = r + half_ * 8;
      int gi = itile * 16 + m;
      if (gi < SEQ) {
        float val = acc[r] / rowsum[m];
        attn_out[((size_t)b * SEQ + gi) * DIM + h * HDIM + wave * 16 + n_] =
            (bf16)val;
      }
    }
  }
}

// ---------------------------------------------------------------------------
// Output projection + bias + exact GELU: [8224x512] @ [512x512] -> f32 out
// one wave per 16x64 tile (4 accumulators; A fragment reused 4x)
// ---------------------------------------------------------------------------
__global__ __launch_bounds__(32) void out_gemm(
    const bf16* __restrict__ ab,   // [8224,512]
    const bf16* __restrict__ wT,   // [512,512] n-major
    const float* __restrict__ bias, float* __restrict__ out) {
  int mt = blockIdx.x, nt4 = blockIdx.y, lane = threadIdx.x;
  v8f acc0 = {0.f, 0.f, 0.f, 0.f, 0.f, 0.f, 0.f, 0.f};
  v8f acc1 = acc0, acc2 = acc0, acc3 = acc0;
#pragma unroll 2
  for (int kk = 0; kk < 16; ++kk) {
    v16bf a = load_A(ab + (size_t)mt * 16 * DIM + kk * 32, DIM, lane);
    const bf16* wbase = wT + (size_t)nt4 * 64 * DIM + kk * 32;
    v16bf b0 = load_B_nmajor(wbase + (size_t)0 * 16 * DIM, DIM, lane);
    acc0 = wmma_bf16(a, b0, acc0);
    v16bf b1 = load_B_nmajor(wbase + (size_t)1 * 16 * DIM, DIM, lane);
    acc1 = wmma_bf16(a, b1, acc1);
    v16bf b2 = load_B_nmajor(wbase + (size_t)2 * 16 * DIM, DIM, lane);
    acc2 = wmma_bf16(a, b2, acc2);
    v16bf b3 = load_B_nmajor(wbase + (size_t)3 * 16 * DIM, DIM, lane);
    acc3 = wmma_bf16(a, b3, acc3);
  }
  int n = lane & 15, half = lane >> 4;
  v8f accs[4] = {acc0, acc1, acc2, acc3};
#pragma unroll
  for (int t = 0; t < 4; ++t) {
    int c = nt4 * 64 + t * 16 + n;
    float bc = bias[c];
#pragma unroll
    for (int r = 0; r < 8; ++r) {
      int m = r + half * 8;
      int gm = mt * 16 + m;
      float v = accs[t][r] + bc;
      v = 0.5f * v * (1.0f + erff(v * 0.70710678118f)); // exact GELU
      out[(size_t)gm * DIM + c] = v;
    }
  }
}

// ---------------------------------------------------------------------------
// Host launch
// ---------------------------------------------------------------------------
extern "C" void kernel_launch(void* const* d_in, const int* in_sizes, int n_in,
                              void* d_out, int out_size, void* d_ws,
                              size_t ws_size, hipStream_t stream) {
  const float*         x       = (const float*)d_in[0];
  const unsigned char* mask    = (const unsigned char*)d_in[1]; // jnp bool
  const int*           rel_pos = (const int*)d_in[2];
  const float*         w_qkv   = (const float*)d_in[3];
  const float*         rel_emb = (const float*)d_in[4];
  const float*         rel_mul = (const float*)d_in[5];
  const float*         w_out   = (const float*)d_in[6];
  const float*         b_out   = (const float*)d_in[7];
  float*               out     = (float*)d_out;

  // carve workspace (~47 MB)
  char* base = (char*)d_ws;
  size_t off = 0;
  auto carve = [&](size_t bytes) -> void* {
    void* p = base + off;
    off += (bytes + 255) & ~(size_t)255;
    return p;
  };
  bf16* xb     = (bf16*)carve((size_t)MROWS * DIM * 2);            // 8.4 MB
  bf16* wqkvT  = (bf16*)carve((size_t)3 * DIM * DIM * 2);          // 1.6 MB
  bf16* woutT  = (bf16*)carve((size_t)DIM * DIM * 2);              // 0.5 MB
  bf16* qb     = (bf16*)carve((size_t)BATCH * HEADS * NPAD * HDIM * 2);
  bf16* kb     = (bf16*)carve((size_t)BATCH * HEADS * NPAD * HDIM * 2);
  bf16* vb     = (bf16*)carve((size_t)BATCH * HEADS * NPAD * HDIM * 2);
  bf16* relh   = (bf16*)carve((size_t)HEADS * RPAD * HDIM * 2);    // 1.0 MB
  bf16* attnb  = (bf16*)carve((size_t)MROWS * DIM * 2);            // 8.4 MB
  (void)ws_size; (void)n_in; (void)in_sizes; (void)out_size;

  // 1) precision conversions + padding init
  {
    int n = MROWS * DIM;
    cvt_f32_bf16<<<(n + 255) / 256, 256, 0, stream>>>(x, xb, n);
  }
  {
    int n = DIM * 3 * DIM;
    cvt_transpose<<<(n + 255) / 256, 256, 0, stream>>>(w_qkv, wqkvT, DIM,
                                                       3 * DIM);
  }
  {
    int n = DIM * DIM;
    cvt_transpose<<<(n + 255) / 256, 256, 0, stream>>>(w_out, woutT, DIM, DIM);
  }
  {
    int n = HEADS * RPAD * HDIM;
    cvt_rel<<<(n + 255) / 256, 256, 0, stream>>>(rel_emb, relh);
  }
  {
    int n = BATCH * HEADS * (NPAD - SEQ) * HDIM;
    zero_pad_qkv<<<(n + 255) / 256, 256, 0, stream>>>(qb, kb, vb);
  }

  // 2) QKV projection GEMM (WMMA bf16, 16x64 per wave)
  qkv_gemm<<<dim3(MT, 3 * DIM / 64), 32, 0, stream>>>(xb, wqkvT, qb, kb, vb);

  // 3) fused attention (WMMA proj-bias + scores + softmax + P@V)
  attn_fused<<<dim3(NPAD / 16 - 1 /*17*/, HEADS, BATCH), 128, 0, stream>>>(
      qb, kb, vb, relh, rel_pos, rel_mul, mask, attnb);

  // 4) output projection + exact GELU (WMMA bf16, 16x64 per wave)
  out_gemm<<<dim3(MT, DIM / 64), 32, 0, stream>>>(attnb, woutT, b_out, out);
}
